// StableAudioAttnProcessor2_0_618475290953
// MI455X (gfx1250) — compile-verified, matrix operand order checked by reading, not run on hardware
//
#include <hip/hip_runtime.h>

typedef __bf16 bf16;
typedef __attribute__((ext_vector_type(8)))  bf16  v8bf;
typedef __attribute__((ext_vector_type(16))) bf16  v16bf;
typedef __attribute__((ext_vector_type(8)))  float v8f;

#define S_LEN   2048
#define DMODEL  1536
#define QKV_LD  2304
#define N_HEADS 24
#define KV_HEADS 6

// CDNA5 async Global->LDS path (ASYNCcnt-tracked), if the toolchain exposes it.
#if defined(__has_builtin)
# if __has_builtin(__builtin_amdgcn_global_load_async_to_lds_b128)
#  define USE_ASYNC_LDS 1
# endif
#endif
#ifndef USE_ASYNC_LDS
# define USE_ASYNC_LDS 0
#endif

#if USE_ASYNC_LDS
typedef int vint4 __attribute__((vector_size(4 * sizeof(int))));
typedef __attribute__((address_space(1))) vint4 g_vint4;   // global
typedef __attribute__((address_space(3))) vint4 l_vint4;   // LDS
static __device__ __forceinline__ void async_cp16(const bf16* gsrc, bf16* ldst) {
  __builtin_amdgcn_global_load_async_to_lds_b128((g_vint4*)gsrc, (l_vint4*)ldst, 0, 0);
}
# if defined(__has_builtin) && __has_builtin(__builtin_amdgcn_s_wait_asynccnt)
#  define ASYNC_WAIT(n) __builtin_amdgcn_s_wait_asynccnt(n)
# else
#  define ASYNC_WAIT(n) asm volatile("s_wait_asynccnt %0" :: "i"(n) : "memory")
# endif
#endif

// ---------- helpers ----------
static __device__ __forceinline__ bf16 f2bf(float f) {
  unsigned u = __builtin_bit_cast(unsigned, f);
  unsigned r = u + 0x7FFFu + ((u >> 16) & 1u);
  unsigned short h = (unsigned short)(r >> 16);
  return __builtin_bit_cast(bf16, h);
}
static __device__ __forceinline__ float bf2f(bf16 h) {
  unsigned short s = __builtin_bit_cast(unsigned short, h);
  unsigned u = ((unsigned)s) << 16;
  return __builtin_bit_cast(float, u);
}

static __device__ __forceinline__ v8f wmma_bf16(v16bf a, v16bf b, v8f c) {
  return __builtin_amdgcn_wmma_f32_16x16x32_bf16(false, a, false, b, (short)0, c,
                                                 false, false);
}

// 16x32 bf16 A/B fragment load from row-major memory (CDNA5 wave32 layout):
// lanes 0-15 : row = lane,    halves 0..7 = K[0..7],  halves 8..15 = K[16..23]
// lanes 16-31: row = lane-16, halves 0..7 = K[8..15], halves 8..15 = K[24..31]
static __device__ __forceinline__ v16bf load_frag(const bf16* base, long ld, int lane) {
  int r  = lane & 15;
  int ks = (lane >> 4) << 3;               // 0 or 8
  const bf16* p = base + (long)r * ld + ks;
  v8bf lo = *(const v8bf*)(p);
  v8bf hi = *(const v8bf*)(p + 16);
  return __builtin_shufflevector(lo, hi, 0,1,2,3,4,5,6,7,8,9,10,11,12,13,14,15);
}

template<int XORM>
static __device__ __forceinline__ float lane_xor_f(float x) {
  int i = __builtin_bit_cast(int, x);
  i = __builtin_amdgcn_ds_swizzle(i, (XORM << 10) | 0x1f);  // group-of-32 xor swizzle
  return __builtin_bit_cast(float, i);
}

// ---------- conversion / transpose kernels ----------
__global__ __launch_bounds__(256) void xcvt_kernel(const float* __restrict__ X,
                                                   bf16* __restrict__ Xb, long n) {
  long i = (long)blockIdx.x * blockDim.x + threadIdx.x;
  if (i < n) Xb[i] = f2bf(X[i]);
}

// W: [R][C] fp32 row-major  ->  Wt: [C][1536] bf16 (Wt[c][r] = W[r][c]); R == 1536
__global__ __launch_bounds__(256) void wtrans_kernel(const float* __restrict__ W,
                                                     bf16* __restrict__ Wt,
                                                     int R, int C) {
  __shared__ float tile[32][33];
  int r0 = blockIdx.x * 32, c0 = blockIdx.y * 32;
  int tx = threadIdx.x & 31, ty = threadIdx.x >> 5;
  (void)R;
#pragma unroll
  for (int i = ty; i < 32; i += 8)
    tile[i][tx] = W[(long)(r0 + i) * C + (c0 + tx)];
  __syncthreads();
#pragma unroll
  for (int i = ty; i < 32; i += 8)
    Wt[(long)(c0 + i) * 1536 + (r0 + tx)] = f2bf(tile[tx][i]);
}

// V columns of QKV -> VT[(b*6+kh)*64 + d][token]
__global__ __launch_bounds__(256) void vtrans_kernel(const bf16* __restrict__ QKV,
                                                     bf16* __restrict__ VT) {
  __shared__ bf16 tile[32][33];
  int b  = blockIdx.x / KV_HEADS;
  int kh = blockIdx.x % KV_HEADS;
  int t0 = blockIdx.y * 32;
  int d0 = blockIdx.z * 32;
  int tx = threadIdx.x & 31, ty = threadIdx.x >> 5;
#pragma unroll
  for (int i = ty; i < 32; i += 8)
    tile[i][tx] = QKV[(long)(b * S_LEN + t0 + i) * QKV_LD + 1920 + kh * 64 + d0 + tx];
  __syncthreads();
#pragma unroll
  for (int i = ty; i < 32; i += 8)
    VT[((long)(b * KV_HEADS + kh) * 64 + d0 + i) * S_LEN + (t0 + tx)] = tile[tx][i];
}

// ---------- partial RoPE (rot_dim = 32) on Q and K heads of QKV, in place ----------
__global__ __launch_bounds__(256) void rope_kernel(bf16* __restrict__ QKV,
                                                   const float* __restrict__ cosT,
                                                   const float* __restrict__ sinT) {
  long idx = (long)blockIdx.x * blockDim.x + threadIdx.x;  // 4096*30*16 threads
  int  i   = (int)(idx & 15);
  long tmp = idx >> 4;
  int  hh  = (int)(tmp % 30);
  long bt  = tmp / 30;
  int  s   = (int)(bt & (S_LEN - 1));
  bf16* base = QKV + bt * QKV_LD + (hh < N_HEADS ? hh * 64 : 1536 + (hh - N_HEADS) * 64);
  float x0 = bf2f(base[i]);
  float x1 = bf2f(base[i + 16]);
  float c0 = cosT[s * 32 + i],      sn0 = sinT[s * 32 + i];
  float c1 = cosT[s * 32 + i + 16], sn1 = sinT[s * 32 + i + 16];
  base[i]      = f2bf(x0 * c0 - x1 * sn0);
  base[i + 16] = f2bf(x1 * c1 + x0 * sn1);
}

// ---------- bf16 WMMA GEMM: C[M][N] = A[M][K] * Bt[N][K]^T, bf16 out ----------
__global__ __launch_bounds__(256) void gemm_bf16_kernel(const bf16* __restrict__ A,
                                                        const bf16* __restrict__ Bt,
                                                        bf16* __restrict__ C,
                                                        int M, int N, int K) {
  int lane = threadIdx.x & 31;
  int wave = threadIdx.x >> 5;
  int m0 = blockIdx.x * 128 + (wave >> 1) * 32;
  int n0 = blockIdx.y * 128 + (wave & 1) * 64;
  v8f acc[2][4] = {};
  for (int k = 0; k < K; k += 32) {
    v16bf a0 = load_frag(A + (long)m0 * K + k,        K, lane);
    v16bf a1 = load_frag(A + (long)(m0 + 16) * K + k, K, lane);
#pragma unroll
    for (int t = 0; t < 4; t++) {
      v16bf b = load_frag(Bt + (long)(n0 + 16 * t) * K + k, K, lane);
      acc[0][t] = wmma_bf16(a0, b, acc[0][t]);
      acc[1][t] = wmma_bf16(a1, b, acc[1][t]);
    }
  }
  int half8 = (lane >> 4) << 3;
#pragma unroll
  for (int i = 0; i < 2; i++)
#pragma unroll
    for (int t = 0; t < 4; t++)
#pragma unroll
      for (int r = 0; r < 8; r++) {
        int row = m0 + 16 * i + r + half8;
        int col = n0 + 16 * t + (lane & 15);
        C[(long)row * N + col] = f2bf(acc[i][t][r]);
      }
}

// ---------- bf16 WMMA GEMM with fp32 epilogue: out = A*Bt^T + bias + resid ----------
__global__ __launch_bounds__(256) void gemm_out_kernel(const bf16* __restrict__ A,
                                                       const bf16* __restrict__ Bt,
                                                       const float* __restrict__ resid,
                                                       const float* __restrict__ bias,
                                                       float* __restrict__ Cout,
                                                       int M, int N, int K) {
  int lane = threadIdx.x & 31;
  int wave = threadIdx.x >> 5;
  int m0 = blockIdx.x * 128 + (wave >> 1) * 32;
  int n0 = blockIdx.y * 128 + (wave & 1) * 64;
  v8f acc[2][4] = {};
  for (int k = 0; k < K; k += 32) {
    v16bf a0 = load_frag(A + (long)m0 * K + k,        K, lane);
    v16bf a1 = load_frag(A + (long)(m0 + 16) * K + k, K, lane);
#pragma unroll
    for (int t = 0; t < 4; t++) {
      v16bf b = load_frag(Bt + (long)(n0 + 16 * t) * K + k, K, lane);
      acc[0][t] = wmma_bf16(a0, b, acc[0][t]);
      acc[1][t] = wmma_bf16(a1, b, acc[1][t]);
    }
  }
  int half8 = (lane >> 4) << 3;
#pragma unroll
  for (int i = 0; i < 2; i++)
#pragma unroll
    for (int t = 0; t < 4; t++)
#pragma unroll
      for (int r = 0; r < 8; r++) {
        int row = m0 + 16 * i + r + half8;
        int col = n0 + 16 * t + (lane & 15);
        long off = (long)row * N + col;
        Cout[off] = acc[i][t][r] + bias[col] + resid[off];
      }
}

// ---------- flash attention ----------
// Block = 8 waves, all sharing (b,h): K/V 32-key tiles are staged cooperatively
// into LDS once per block (async Global->LDS double-buffered when available),
// then every wave reads its WMMA fragments from LDS instead of 8x-redundant VMEM.
#define KT_LD 72   // 64 elems + pad, 144B rows (16B aligned)
#define VT_LD 40   // 32 elems + pad,  80B rows (16B aligned)

__global__ __launch_bounds__(256) void attn_kernel(const bf16* __restrict__ QKV,
                                                   const bf16* __restrict__ VT,
                                                   bf16* __restrict__ CTX) {
#if USE_ASYNC_LDS
  __shared__ __align__(16) bf16 kTls[2][32 * KT_LD];
  __shared__ __align__(16) bf16 vTls[2][64 * VT_LD];
#else
  __shared__ __align__(16) bf16 kTls[1][32 * KT_LD];
  __shared__ __align__(16) bf16 vTls[1][64 * VT_LD];
#endif
  __shared__ __align__(16) bf16 pTile[8][16 * 32];

  int tid  = threadIdx.x;
  int lane = tid & 31;
  int wave = tid >> 5;
  int b  = blockIdx.x / N_HEADS;
  int h  = blockIdx.x % N_HEADS;
  int kh = h >> 2;                       // GQA: rep = 4
  int q0 = blockIdx.y * 128 + wave * 16;

  const bf16* Qb = QKV + (long)(b * S_LEN + q0) * QKV_LD + h * 64;
  const bf16* Kg = QKV + (long)(b * S_LEN) * QKV_LD + 1536 + kh * 64;
  const bf16* Vg = VT + (long)((b * KV_HEADS + kh) * 64) * S_LEN;
  bf16* myP = pTile[wave];

  // one 16B chunk of K and one of V per thread per tile
  int skey = tid >> 3, sdo = (tid & 7) * 8;      // K: 32 rows x 8 chunks
  int sd   = tid >> 2, sko = (tid & 3) * 8;      // V: 64 rows x 4 chunks
  const bf16* kSrc = Kg + (long)skey * QKV_LD + sdo;
  const bf16* vSrc = Vg + (long)sd * S_LEN + sko;

  v16bf qa0 = load_frag(Qb,      QKV_LD, lane);  // d = 0..31
  v16bf qa1 = load_frag(Qb + 32, QKV_LD, lane);  // d = 32..63

  v8f o[4] = {};
  float mrow[8], lrow[8];
#pragma unroll
  for (int r = 0; r < 8; r++) { mrow[r] = -1e30f; lrow[r] = 0.f; }
  const float scale = 0.125f;                    // 1/sqrt(64)

#if USE_ASYNC_LDS
  // prologue: async-stage tile 0 into buffer 0
  async_cp16(kSrc, &kTls[0][skey * KT_LD + sdo]);
  async_cp16(vSrc, &vTls[0][sd * VT_LD + sko]);
#endif
  int p = 0;

  for (int kt = 0; kt < S_LEN; kt += 32) {
#if USE_ASYNC_LDS
    bool more = (kt + 32) < S_LEN;
    if (more) {                                  // stage next tile into other buffer
      async_cp16(kSrc + (long)(kt + 32) * QKV_LD, &kTls[p ^ 1][skey * KT_LD + sdo]);
      async_cp16(vSrc + kt + 32,                  &vTls[p ^ 1][sd * VT_LD + sko]);
      ASYNC_WAIT(2);                             // current tile's 2 ops retired
    } else {
      ASYNC_WAIT(0);
    }
#else
    {                                            // VGPR-staged fallback
      uint4 kv = *(const uint4*)(kSrc + (long)kt * QKV_LD);
      uint4 vv = *(const uint4*)(vSrc + kt);
      *(uint4*)(&kTls[0][skey * KT_LD + sdo]) = kv;
      *(uint4*)(&vTls[0][sd * VT_LD + sko])   = vv;
    }
#endif
    __syncthreads();                             // tile visible to all waves
    const bf16* kbuf = kTls[p];
    const bf16* vbuf = vTls[p];

    // scores: two 16x16 tiles (keys kt..kt+15, kt+16..kt+31)
    v8f s0 = {}, s1 = {};
    s0 = wmma_bf16(qa0, load_frag(kbuf,                 KT_LD, lane), s0);
    s0 = wmma_bf16(qa1, load_frag(kbuf + 32,            KT_LD, lane), s0);
    s1 = wmma_bf16(qa0, load_frag(kbuf + 16 * KT_LD,      KT_LD, lane), s1);
    s1 = wmma_bf16(qa1, load_frag(kbuf + 16 * KT_LD + 32, KT_LD, lane), s1);

    // online softmax per row (row M = r + 8*(lane/16), N striped across 16 lanes)
    int half8 = (lane >> 4) << 3;
#pragma unroll
    for (int r = 0; r < 8; r++) {
      float v0 = s0[r] * scale, v1 = s1[r] * scale;
      float mx = fmaxf(v0, v1);
      mx = fmaxf(mx, lane_xor_f<1>(mx));
      mx = fmaxf(mx, lane_xor_f<2>(mx));
      mx = fmaxf(mx, lane_xor_f<4>(mx));
      mx = fmaxf(mx, lane_xor_f<8>(mx));
      float mnew = fmaxf(mrow[r], mx);
      float p0 = __expf(v0 - mnew), p1 = __expf(v1 - mnew);
      float ps = p0 + p1;
      ps += lane_xor_f<1>(ps);
      ps += lane_xor_f<2>(ps);
      ps += lane_xor_f<4>(ps);
      ps += lane_xor_f<8>(ps);
      float alpha = __expf(mrow[r] - mnew);
      lrow[r] = lrow[r] * alpha + ps;
      mrow[r] = mnew;
      o[0][r] *= alpha; o[1][r] *= alpha; o[2][r] *= alpha; o[3][r] *= alpha;
      // C-layout -> row-major P tile in LDS (bf16 16x32)
      int M = r + half8;
      int N = lane & 15;
      myP[M * 32 + N]      = f2bf(p0);
      myP[M * 32 + N + 16] = f2bf(p1);
    }
    asm volatile("s_wait_dscnt 0" ::: "memory");   // wave-private LDS RAW fence
    v16bf pa = load_frag(myP, 32, lane);           // re-read as A fragment

    // O += P(16x32) @ V(32x64)
#pragma unroll
    for (int t = 0; t < 4; t++) {
      v16bf vb = load_frag(vbuf + t * 16 * VT_LD, VT_LD, lane);
      o[t] = wmma_bf16(pa, vb, o[t]);
    }
    __syncthreads();                             // all waves done with buffer p
#if USE_ASYNC_LDS
    p ^= 1;
#endif
  }

  int half8 = (lane >> 4) << 3;
#pragma unroll
  for (int t = 0; t < 4; t++)
#pragma unroll
    for (int r = 0; r < 8; r++) {
      int row = q0 + r + half8;
      int col = h * 64 + t * 16 + (lane & 15);
      CTX[(long)(b * S_LEN + row) * DMODEL + col] = f2bf(o[t][r] * (1.f / lrow[r]));
    }
}

// ---------- host ----------
extern "C" void kernel_launch(void* const* d_in, const int* in_sizes, int n_in,
                              void* d_out, int out_size, void* d_ws, size_t ws_size,
                              hipStream_t stream) {
  (void)in_sizes; (void)n_in; (void)out_size; (void)ws_size;
  const float* X    = (const float*)d_in[0];
  const float* cosT = (const float*)d_in[1];
  const float* sinT = (const float*)d_in[2];
  const float* Wq   = (const float*)d_in[3];
  const float* Wk   = (const float*)d_in[4];
  const float* Wv   = (const float*)d_in[5];
  const float* Wo   = (const float*)d_in[6];
  const float* bo   = (const float*)d_in[7];
  float* out = (float*)d_out;

  char* ws = (char*)d_ws;                 // ~59 MB total, 256B-aligned sections
  bf16* Xbf   = (bf16*)(ws + 0);          // 4096*1536       = 12,582,912 B
  bf16* WqkvT = (bf16*)(ws + 12582912);   // 2304*1536       =  7,077,888 B
  bf16* WoT   = (bf16*)(ws + 19660800);   // 1536*1536       =  4,718,592 B
  bf16* QKV   = (bf16*)(ws + 24379392);   // 4096*2304       = 18,874,368 B
  bf16* VT    = (bf16*)(ws + 43253760);   // 12*64*2048      =  3,145,728 B
  bf16* CTX   = (bf16*)(ws + 46399488);   // 4096*1536       = 12,582,912 B

  xcvt_kernel<<<24576, 256, 0, stream>>>(X, Xbf, 6291456L);
  wtrans_kernel<<<dim3(48, 48), 256, 0, stream>>>(Wq, WqkvT,               1536, 1536);
  wtrans_kernel<<<dim3(48, 12), 256, 0, stream>>>(Wk, WqkvT + 1536 * 1536, 1536, 384);
  wtrans_kernel<<<dim3(48, 12), 256, 0, stream>>>(Wv, WqkvT + 1920 * 1536, 1536, 384);
  wtrans_kernel<<<dim3(48, 48), 256, 0, stream>>>(Wo, WoT,                 1536, 1536);
  gemm_bf16_kernel<<<dim3(32, 18), 256, 0, stream>>>(Xbf, WqkvT, QKV, 4096, 2304, 1536);
  rope_kernel<<<7680, 256, 0, stream>>>(QKV, cosT, sinT);
  vtrans_kernel<<<dim3(12, 64, 2), 256, 0, stream>>>(QKV, VT);
  attn_kernel<<<dim3(48, 16), 256, 0, stream>>>(QKV, VT, CTX);
  gemm_out_kernel<<<dim3(32, 12), 256, 0, stream>>>(CTX, WoT, X, bo, out, 4096, 1536, 1536);
}
